// Attention_79216376808241
// MI455X (gfx1250) — matrix operand order repaired and checked by reading, not compiled, not run on hardware
//
#include <hip/hip_runtime.h>
#include <hip/hip_bf16.h>
#include <math.h>

#define B_DIM 2
#define S_DIM 2048
#define H_DIM 2048
#define NH    16
#define NKV   4
#define HD    128
#define NREP  (NH / NKV)

typedef __attribute__((ext_vector_type(16))) __bf16 v16bf;
typedef __attribute__((ext_vector_type(8)))  float  v8f;

struct U128 { unsigned int w[4]; };

union BF16x16 {
    v16bf v;
    U128  q[2];
    __bf16 e[16];
};

// ---------------------------------------------------------------------------
// CDNA5 async global->LDS copy (ASYNCcnt-tracked) + wait
__device__ __forceinline__ void async_load_b128(const void* gaddr, void* lds_ptr) {
    // Generic address of a __shared__ object: low 32 bits are the LDS byte offset
    unsigned lds_off = (unsigned)(size_t)lds_ptr;
    asm volatile("global_load_async_to_lds_b128 %0, %1, off"
                 :: "v"(lds_off), "v"(gaddr)
                 : "memory");
}
__device__ __forceinline__ void wait_async0() {
#if __has_builtin(__builtin_amdgcn_s_wait_asynccnt)
    __builtin_amdgcn_s_wait_asynccnt(0);
#else
    asm volatile("s_wait_asynccnt 0x0" ::: "memory");
#endif
}

// ---------------------------------------------------------------------------
// Fragment loaders (layouts per CDNA5 ISA 7.12.2)
// A-matrix 16x32 bf16: lane half=0 -> K {0..7,16..23}, half=1 -> K {8..15,24..31}
__device__ __forceinline__ v16bf load_frag_a(const __bf16* p, int half) {
    BF16x16 f;
    const __bf16* base = p + half * 8;
    f.q[0] = *reinterpret_cast<const U128*>(base);
    f.q[1] = *reinterpret_cast<const U128*>(base + 16);
    return f.v;
}
// B-matrix 32x16 bf16 stored transposed (row = N, 32 contiguous K):
__device__ __forceinline__ v16bf load_frag_b(const __bf16* p, int half) {
    BF16x16 f;
    const __bf16* base = p + half * 8;
    f.q[0] = *reinterpret_cast<const U128*>(base);
    f.q[1] = *reinterpret_cast<const U128*>(base + 16);
    return f.v;
}

__device__ __forceinline__ void store_c(float* p, float v)  { *p = v; }
__device__ __forceinline__ void store_c(__bf16* p, float v) { *p = (__bf16)v; }

// ---------------------------------------------------------------------------
// Per-thread staging registers for the A tile (global -> regs -> LDS pipeline)
template <typename AT> struct ARegs;
template <> struct ARegs<float>  { float4 f[4]; };
template <> struct ARegs<__bf16> { U128   u[2]; };

__device__ __forceinline__ void load_a_regs(ARegs<float>& r, const float* A,
                                            int K, int mb, int k0, int tid) {
#pragma unroll
    for (int i = 0; i < 4; ++i) {
        int idx = tid + i * 256;          // 128 rows * 8 float4
        int row = idx >> 3;
        int c4  = (idx & 7) * 4;
        r.f[i] = *reinterpret_cast<const float4*>(A + (size_t)(mb + row) * K + k0 + c4);
    }
}
__device__ __forceinline__ void store_a_lds(const ARegs<float>& r,
                                            __bf16 (*As)[40], int tid) {
#pragma unroll
    for (int i = 0; i < 4; ++i) {
        int idx = tid + i * 256;
        int row = idx >> 3;
        int c4  = (idx & 7) * 4;
        As[row][c4 + 0] = (__bf16)r.f[i].x;
        As[row][c4 + 1] = (__bf16)r.f[i].y;
        As[row][c4 + 2] = (__bf16)r.f[i].z;
        As[row][c4 + 3] = (__bf16)r.f[i].w;
    }
}
__device__ __forceinline__ void load_a_regs(ARegs<__bf16>& r, const __bf16* A,
                                            int K, int mb, int k0, int tid) {
#pragma unroll
    for (int i = 0; i < 2; ++i) {
        int idx = tid + i * 256;          // 128 rows * 4 chunks of 8 bf16
        int row = idx >> 2;
        int c8  = (idx & 3) * 8;
        r.u[i] = *reinterpret_cast<const U128*>(A + (size_t)(mb + row) * K + k0 + c8);
    }
}
__device__ __forceinline__ void store_a_lds(const ARegs<__bf16>& r,
                                            __bf16 (*As)[40], int tid) {
#pragma unroll
    for (int i = 0; i < 2; ++i) {
        int idx = tid + i * 256;
        int row = idx >> 2;
        int c8  = (idx & 3) * 8;
        *reinterpret_cast<U128*>(&As[row][c8]) = r.u[i];
    }
}

// ---------------------------------------------------------------------------
// Generic bf16-WMMA GEMM: C[M,N] = A[M,K] * B[K,N] (B fp32 row-major).
// 128x128 block, K-step 32, 8 waves (4x2), wave tile 32x64.
// Global->reg->LDS pipelining across K tiles; fragment loads pipelined by 1.
template <typename AT, typename CT>
__global__ __launch_bounds__(256)
void gemm_wmma(const AT* __restrict__ A, const float* __restrict__ Bw,
               CT* __restrict__ C, int M, int N, int K) {
    __shared__ __bf16 As [128][40];   // [m][k]
    __shared__ __bf16 Bst[128][40];   // [n][k] (transposed)

    const int tid  = threadIdx.x;
    const int lane = tid & 31;
    const int wid  = tid >> 5;
    const int half = lane >> 4;
    const int l15  = lane & 15;
    const int wm   = wid >> 1;
    const int wn   = wid & 1;
    const int mb   = blockIdx.y * 128;
    const int nb   = blockIdx.x * 128;

    v8f acc[2][4];
#pragma unroll
    for (int fm = 0; fm < 2; ++fm)
#pragma unroll
        for (int fn = 0; fn < 4; ++fn)
            acc[fm][fn] = (v8f){0.f, 0.f, 0.f, 0.f, 0.f, 0.f, 0.f, 0.f};

    ARegs<AT> ar;
    float4 br[4];
    load_a_regs(ar, A, K, mb, 0, tid);
#pragma unroll
    for (int i = 0; i < 4; ++i) {
        int idx = tid + i * 256;          // 32 rows * 32 float4
        int row = idx >> 5;
        int c4  = (idx & 31) * 4;
        br[i] = *reinterpret_cast<const float4*>(Bw + (size_t)row * N + nb + c4);
    }

    for (int k0 = 0; k0 < K; k0 += 32) {
        __syncthreads();
        store_a_lds(ar, As, tid);
#pragma unroll
        for (int i = 0; i < 4; ++i) {
            int idx = tid + i * 256;
            int row = idx >> 5;
            int c4  = (idx & 31) * 4;
            Bst[c4 + 0][row] = (__bf16)br[i].x;
            Bst[c4 + 1][row] = (__bf16)br[i].y;
            Bst[c4 + 2][row] = (__bf16)br[i].z;
            Bst[c4 + 3][row] = (__bf16)br[i].w;
        }
        __syncthreads();

        // Overlap next tile's global loads with this tile's WMMAs
        if (k0 + 32 < K) {
            load_a_regs(ar, A, K, mb, k0 + 32, tid);
#pragma unroll
            for (int i = 0; i < 4; ++i) {
                int idx = tid + i * 256;
                int row = idx >> 5;
                int c4  = (idx & 31) * 4;
                br[i] = *reinterpret_cast<const float4*>(
                    Bw + (size_t)(k0 + 32 + row) * N + nb + c4);
            }
        }

        v16bf a0 = load_frag_a(&As[wm * 32 + l15][0],      half);
        v16bf a1 = load_frag_a(&As[wm * 32 + 16 + l15][0], half);
        v16bf bcur = load_frag_b(&Bst[wn * 64 + l15][0], half);
#pragma unroll
        for (int fn = 0; fn < 4; ++fn) {
            v16bf bnext = bcur;
            if (fn < 3)
                bnext = load_frag_b(&Bst[wn * 64 + (fn + 1) * 16 + l15][0], half);
            acc[0][fn] = __builtin_amdgcn_wmma_f32_16x16x32_bf16(
                false, a0, false, bcur, (short)0, acc[0][fn], false, false);
            acc[1][fn] = __builtin_amdgcn_wmma_f32_16x16x32_bf16(
                false, a1, false, bcur, (short)0, acc[1][fn], false, false);
            bcur = bnext;
        }
    }

#pragma unroll
    for (int fm = 0; fm < 2; ++fm)
#pragma unroll
        for (int fn = 0; fn < 4; ++fn)
#pragma unroll
            for (int r = 0; r < 8; ++r) {
                int row = mb + wm * 32 + fm * 16 + half * 8 + r;
                int col = nb + wn * 64 + fn * 16 + l15;
                store_c(&C[(size_t)row * N + col], acc[fm][fn][r]);
            }
}

// ---------------------------------------------------------------------------
// In-place RoPE on a bf16 [B, S, nheads, HD] tensor.
__global__ __launch_bounds__(256)
void rope_kernel(__bf16* __restrict__ t, const float* __restrict__ cosT,
                 const float* __restrict__ sinT, int nheads, int total) {
    int tid = blockIdx.x * blockDim.x + threadIdx.x;
    if (tid >= total) return;
    int i  = tid & 63;
    int hh = (tid >> 6) % nheads;
    int s  = (tid / (64 * nheads)) % S_DIM;
    int b  = tid / (64 * nheads * S_DIM);
    size_t base = ((size_t)(b * S_DIM + s) * nheads + hh) * HD;
    float x0 = (float)t[base + i];
    float x1 = (float)t[base + i + 64];
    float c0 = cosT[s * HD + i],      s0 = sinT[s * HD + i];
    float c1 = cosT[s * HD + i + 64], s1 = sinT[s * HD + i + 64];
    t[base + i]      = (__bf16)(x0 * c0 - x1 * s0);
    t[base + i + 64] = (__bf16)(x1 * c1 + x0 * s1);
}

// ---------------------------------------------------------------------------
// Flash attention, causal, GQA. Grid: (S/128, NH, B). 8 waves; wave w owns
// q rows w*16..w*16+15 across all 128 k-cols (softmax stats half-wave-local).
__global__ __launch_bounds__(256)
void flash_attn(const __bf16* __restrict__ q, const __bf16* __restrict__ k,
                const __bf16* __restrict__ v, __bf16* __restrict__ o) {
    __shared__ __bf16 Ks [128][136];   // [kpos][hd]
    __shared__ __bf16 Vst[128][136];   // [hd][kpos]
    __shared__ __bf16 Ps [128][136];   // [qrow][kpos] wave-private rows

    const int qt   = blockIdx.x;
    const int h    = blockIdx.y;
    const int b    = blockIdx.z;
    const int kvh  = h / NREP;
    const int tid  = threadIdx.x;
    const int lane = tid & 31;
    const int wid  = tid >> 5;
    const int half = lane >> 4;
    const int l15  = lane & 15;
    const float scale = 0.08838834764831845f;   // 1/sqrt(HD)

    v16bf qf[4];
    {
        const int qrow = qt * 128 + wid * 16 + l15;
        const __bf16* qp = q + ((size_t)(b * S_DIM + qrow) * NH + h) * HD;
#pragma unroll
        for (int kc = 0; kc < 4; ++kc) qf[kc] = load_frag_a(qp + kc * 32, half);
    }

    v8f O[8];
#pragma unroll
    for (int nf = 0; nf < 8; ++nf) O[nf] = (v8f){0.f,0.f,0.f,0.f,0.f,0.f,0.f,0.f};
    v8f m, lsum;
#pragma unroll
    for (int r = 0; r < 8; ++r) { m[r] = -INFINITY; lsum[r] = 0.f; }

    for (int kt = 0; kt <= qt; ++kt) {
        __syncthreads();
        // K tile via CDNA5 async global->LDS; V tile manual (transpose scatter)
#pragma unroll
        for (int i = 0; i < 8; ++i) {
            int idx = tid + i * 256;          // 128 rows * 16 chunks of 8 bf16
            int row = idx >> 4;
            int c8  = (idx & 15) * 8;
            size_t gofs = ((size_t)(b * S_DIM + kt * 128 + row) * NKV + kvh) * HD + c8;
            async_load_b128(k + gofs, &Ks[row][c8]);
            U128 vv = *reinterpret_cast<const U128*>(v + gofs);
            const __bf16* ve = reinterpret_cast<const __bf16*>(&vv);
#pragma unroll
            for (int j = 0; j < 8; ++j) Vst[c8 + j][row] = ve[j];
        }
        wait_async0();
        __syncthreads();

        // S = Q K^T (raw units; scale folded into the exponent later)
        v8f sfr[8];
        {
            v16bf bcur = load_frag_b(&Ks[l15][0], half);
#pragma unroll
            for (int nf = 0; nf < 8; ++nf) {
                v8f s = (v8f){0.f,0.f,0.f,0.f,0.f,0.f,0.f,0.f};
#pragma unroll
                for (int kc = 0; kc < 4; ++kc) {
                    v16bf bnext = bcur;
                    if (!(nf == 7 && kc == 3)) {
                        int nf2 = (kc == 3) ? nf + 1 : nf;
                        int kc2 = (kc == 3) ? 0 : kc + 1;
                        bnext = load_frag_b(&Ks[nf2 * 16 + l15][kc2 * 32], half);
                    }
                    s = __builtin_amdgcn_wmma_f32_16x16x32_bf16(
                            false, qf[kc], false, bcur, (short)0, s, false, false);
                    bcur = bnext;
                }
                sfr[nf] = s;
            }
        }
        // Causal mask (diagonal tile only)
        if (kt == qt) {
#pragma unroll
            for (int nf = 0; nf < 8; ++nf)
#pragma unroll
                for (int r = 0; r < 8; ++r) {
                    int qrow = wid * 16 + half * 8 + r;
                    int kcol = nf * 16 + l15;
                    if (kcol > qrow) sfr[nf][r] = -INFINITY;
                }
        }
        // Row max across frags then across the 16 lanes of this half-wave
        v8f mt = sfr[0];
#pragma unroll
        for (int nf = 1; nf < 8; ++nf)
#pragma unroll
            for (int r = 0; r < 8; ++r) mt[r] = fmaxf(mt[r], sfr[nf][r]);
#pragma unroll
        for (int msk = 1; msk < 16; msk <<= 1)
#pragma unroll
            for (int r = 0; r < 8; ++r) mt[r] = fmaxf(mt[r], __shfl_xor(mt[r], msk, 16));

        v8f alpha;
#pragma unroll
        for (int r = 0; r < 8; ++r) {
            float mn = fmaxf(m[r], mt[r]);
            alpha[r] = __expf((m[r] - mn) * scale);
            m[r] = mn;
        }
        // P = exp((S - m)*scale); row sums
        v8f rs = (v8f){0.f,0.f,0.f,0.f,0.f,0.f,0.f,0.f};
#pragma unroll
        for (int nf = 0; nf < 8; ++nf)
#pragma unroll
            for (int r = 0; r < 8; ++r) {
                float p = __expf((sfr[nf][r] - m[r]) * scale);
                sfr[nf][r] = p;
                rs[r] += p;
            }
#pragma unroll
        for (int msk = 1; msk < 16; msk <<= 1)
#pragma unroll
            for (int r = 0; r < 8; ++r) rs[r] += __shfl_xor(rs[r], msk, 16);
#pragma unroll
        for (int r = 0; r < 8; ++r) lsum[r] = lsum[r] * alpha[r] + rs[r];
#pragma unroll
        for (int nf = 0; nf < 8; ++nf)
#pragma unroll
            for (int r = 0; r < 8; ++r) O[nf][r] *= alpha[r];

        // P: C-layout -> A-layout via wave-private LDS rows
#pragma unroll
        for (int nf = 0; nf < 8; ++nf)
#pragma unroll
            for (int r = 0; r < 8; ++r)
                Ps[wid * 16 + half * 8 + r][nf * 16 + l15] = (__bf16)sfr[nf][r];

        v16bf pf[4];
        const __bf16* pp = &Ps[wid * 16 + l15][0];
#pragma unroll
        for (int kc = 0; kc < 4; ++kc) pf[kc] = load_frag_a(pp + kc * 32, half);

        // O += P V  (B-fragment loads pipelined by 1)
        {
            v16bf bcur = load_frag_b(&Vst[l15][0], half);
#pragma unroll
            for (int nf = 0; nf < 8; ++nf)
#pragma unroll
                for (int kc = 0; kc < 4; ++kc) {
                    v16bf bnext = bcur;
                    if (!(nf == 7 && kc == 3)) {
                        int nf2 = (kc == 3) ? nf + 1 : nf;
                        int kc2 = (kc == 3) ? 0 : kc + 1;
                        bnext = load_frag_b(&Vst[nf2 * 16 + l15][kc2 * 32], half);
                    }
                    O[nf] = __builtin_amdgcn_wmma_f32_16x16x32_bf16(
                                false, pf[kc], false, bcur, (short)0, O[nf], false, false);
                    bcur = bnext;
                }
        }
    }

    v8f rcp;
#pragma unroll
    for (int r = 0; r < 8; ++r) rcp[r] = 1.0f / lsum[r];
    const int qrow0 = qt * 128 + wid * 16;
#pragma unroll
    for (int nf = 0; nf < 8; ++nf)
#pragma unroll
        for (int r = 0; r < 8; ++r) {
            int row = qrow0 + half * 8 + r;
            o[((size_t)(b * S_DIM + row) * NH + h) * HD + nf * 16 + l15] =
                (__bf16)(O[nf][r] * rcp[r]);
        }
}

// ---------------------------------------------------------------------------
extern "C" void kernel_launch(void* const* d_in, const int* in_sizes, int n_in,
                              void* d_out, int out_size, void* d_ws, size_t ws_size,
                              hipStream_t stream) {
    (void)in_sizes; (void)n_in; (void)out_size; (void)ws_size;
    const float* x    = (const float*)d_in[0];
    const float* cosT = (const float*)d_in[1];
    const float* sinT = (const float*)d_in[2];
    const float* wq   = (const float*)d_in[3];
    const float* wk   = (const float*)d_in[4];
    const float* wv   = (const float*)d_in[5];
    const float* wo   = (const float*)d_in[6];
    float* out = (float*)d_out;

    char* ws = (char*)d_ws;
    size_t off = 0;
    auto alloc = [&](size_t bytes) {
        void* p = ws + off;
        off += (bytes + 255) & ~(size_t)255;
        return p;
    };
    __bf16* qb = (__bf16*)alloc((size_t)B_DIM * S_DIM * NH  * HD * 2);
    __bf16* kb = (__bf16*)alloc((size_t)B_DIM * S_DIM * NKV * HD * 2);
    __bf16* vb = (__bf16*)alloc((size_t)B_DIM * S_DIM * NKV * HD * 2);
    __bf16* ab = (__bf16*)alloc((size_t)B_DIM * S_DIM * NH  * HD * 2);

    const int M = B_DIM * S_DIM;     // 4096
    dim3 blk(256);

    gemm_wmma<float, __bf16><<<dim3((NH * HD) / 128, M / 128), blk, 0, stream>>>(
        x, wq, qb, M, NH * HD, H_DIM);
    gemm_wmma<float, __bf16><<<dim3((NKV * HD) / 128, M / 128), blk, 0, stream>>>(
        x, wk, kb, M, NKV * HD, H_DIM);
    gemm_wmma<float, __bf16><<<dim3((NKV * HD) / 128, M / 128), blk, 0, stream>>>(
        x, wv, vb, M, NKV * HD, H_DIM);

    int qtot = B_DIM * S_DIM * NH * (HD / 2);
    rope_kernel<<<qtot / 256, blk, 0, stream>>>(qb, cosT, sinT, NH, qtot);
    int ktot = B_DIM * S_DIM * NKV * (HD / 2);
    rope_kernel<<<ktot / 256, blk, 0, stream>>>(kb, cosT, sinT, NKV, ktot);

    flash_attn<<<dim3(S_DIM / 128, NH, B_DIM), blk, 0, stream>>>(qb, kb, vb, ab);

    gemm_wmma<__bf16, float><<<dim3(H_DIM / 128, M / 128), blk, 0, stream>>>(
        ab, wo, out, M, H_DIM, NH * HD);
}
